// Attention_51969104282059
// MI455X (gfx1250) — matrix operand order repaired and checked by reading, not compiled, not run on hardware
//
#include <hip/hip_runtime.h>

#define NB 8
#define NT 4096
#define NC 1024
#define ND 128

typedef __bf16 bf16;
typedef __attribute__((ext_vector_type(16))) __bf16 v16bf;
typedef __attribute__((ext_vector_type(8)))  __bf16 v8bf;
typedef __attribute__((ext_vector_type(8)))  float  v8f;
typedef __attribute__((ext_vector_type(4)))  float  v4f;

// log2(e) / sqrt(128): folds both the attention scale and the base-2
// exponent conversion into the Q projection (hardware v_exp_f32 is exp2).
#define QSCALE 0.12751751528164f

static __device__ __forceinline__ v8f wmma_bf16(v16bf a, v16bf b, v8f c) {
  return __builtin_amdgcn_wmma_f32_16x16x32_bf16(false, a, false, b, (short)0, c,
                                                 false, false);
}

// ---------------------------------------------------------------------------
// Kernel 0: reformat fp32 weights [C,D] into bf16 WMMA-B-fragment order.
// Fragment layout (16-bit B, 32x16): lane L holds N = L&15, K = (e>>3)*16 + (L>>4)*8 + (e&7).
// Stored so each lane's 16 elems are contiguous: one 32B load per fragment.
// ---------------------------------------------------------------------------
__global__ __launch_bounds__(32) void wrefmt_kernel(
    const float* __restrict__ Wq, const float* __restrict__ Wk,
    const float* __restrict__ Wv, bf16* __restrict__ wbf) {
  int blk = blockIdx.x;            // 0..767
  int m   = blk >> 8;              // 0..2 -> Q,K,V
  int rem = blk & 255;
  int kc  = rem >> 3;              // K-chunk of 32, 0..31
  int nt  = rem & 7;               // N-tile of 16, 0..7
  const float* W = (m == 0) ? Wq : (m == 1) ? Wk : Wv;
  int lane = threadIdx.x;
  int n    = lane & 15;
  int kb   = (lane >> 4) * 8;
  bf16* dst = wbf + ((size_t)((m * 32 + kc) * 8 + nt)) * 512 + lane * 16;
#pragma unroll
  for (int e = 0; e < 16; ++e)
    dst[e] = (bf16)W[(size_t)(kc * 32 + kb + (e & 7) + (e >> 3) * 16) * ND + nt * 16 + n];
}

// ---------------------------------------------------------------------------
// Kernel 1: fused QKV projection. One wave computes one 16-row tile x 128 cols
// for Q, K, V (looped; the 64KB X tile stays hot in WGP$/L2 across passes).
// Q is pre-scaled by log2(e)/sqrt(D). Q,K stored bf16 row-major [B*T, D];
// V stored transposed bf16 [B, D, T] so P@V B-fragments are contiguous loads.
// ---------------------------------------------------------------------------
__global__ __launch_bounds__(128) void qkv_kernel(
    const float* __restrict__ x, const bf16* __restrict__ wbf,
    bf16* __restrict__ qbf, bf16* __restrict__ kbf, bf16* __restrict__ vtbf) {
  int wave = threadIdx.x >> 5;
  int lane = threadIdx.x & 31;
  int tile = blockIdx.x * 4 + wave;   // 0..2047
  int t0   = tile * 16;
  int n    = lane & 15;
  int half = lane >> 4;
  const float* xrow = x + (size_t)(t0 + n) * NC;

#pragma unroll
  for (int m = 0; m < 3; ++m) {
    v8f acc[8];
#pragma unroll
    for (int i = 0; i < 8; ++i)
#pragma unroll
      for (int e = 0; e < 8; ++e) acc[i][e] = 0.0f;

    for (int kc = 0; kc < 32; ++kc) {
      // A fragment (16x32 bf16): lane holds row t0+n, K = (e>>3)*16+half*8+(e&7)
      const float* xp = xrow + kc * 32 + half * 8;
      v4f x0 = *(const v4f*)(xp + 0);
      v4f x1 = *(const v4f*)(xp + 4);
      v4f x2 = *(const v4f*)(xp + 16);
      v4f x3 = *(const v4f*)(xp + 20);
      v16bf a;
#pragma unroll
      for (int e = 0; e < 4; ++e) {
        a[e]      = (bf16)x0[e];
        a[4 + e]  = (bf16)x1[e];
        a[8 + e]  = (bf16)x2[e];
        a[12 + e] = (bf16)x3[e];
      }
      const bf16* wb = wbf + ((size_t)(m * 32 + kc) * 8) * 512 + lane * 16;
#pragma unroll
      for (int nt = 0; nt < 8; ++nt) {
        v16bf bfrag = *(const v16bf*)(wb + (size_t)nt * 512);
        acc[nt] = wmma_bf16(a, bfrag, acc[nt]);
      }
    }

    if (m < 2) {
      const float sc = (m == 0) ? QSCALE : 1.0f;   // fold attn scale into Q
      bf16* dst = (m == 0 ? qbf : kbf) + (size_t)t0 * ND;
#pragma unroll
      for (int r = 0; r < 8; ++r) {
        int M = r + half * 8;
#pragma unroll
        for (int nt = 0; nt < 8; ++nt)
          dst[(size_t)M * ND + nt * 16 + n] = (bf16)(acc[nt][r] * sc);
      }
    } else {
      int b  = t0 >> 12;             // / NT
      int tl = t0 & (NT - 1);
      bf16* dst = vtbf + (size_t)b * ND * NT;
#pragma unroll
      for (int r = 0; r < 8; ++r) {
        int M = r + half * 8;
#pragma unroll
        for (int nt = 0; nt < 8; ++nt)
          dst[(size_t)(nt * 16 + n) * NT + tl + M] = (bf16)acc[nt][r];
      }
    }
  }
}

// ---------------------------------------------------------------------------
// Kernel 2: flash attention, one wave per 16-row q-tile, 32 keys per step.
// Softmax runs in the base-2 domain (scale pre-folded into Q); only blocks
// crossing the causal diagonal pay for masking.
// ---------------------------------------------------------------------------
__global__ __launch_bounds__(128) void attn_kernel(
    const bf16* __restrict__ qbf, const bf16* __restrict__ kbf,
    const bf16* __restrict__ vtbf, float* __restrict__ out) {
  __shared__ __align__(16) bf16 pl[4][16 * 40];   // per-wave P staging, pad=8
  int wave = threadIdx.x >> 5;
  int lane = threadIdx.x & 31;
  int tile = blockIdx.x * 4 + wave;   // 0..2047
  int t0g  = tile * 16;
  int b    = t0g >> 12;
  int t0   = t0g & (NT - 1);
  int n    = lane & 15;
  int half = lane >> 4;
  bf16* plw = pl[wave];

  // Load Q A-fragments (already scaled in projection): pure register packing.
  const bf16* qrow = qbf + (size_t)(t0g + n) * ND;
  v16bf qa[4];
#pragma unroll
  for (int c = 0; c < 4; ++c) {
    v8bf g0 = *(const v8bf*)(qrow + c * 32 + half * 16);
    v8bf g1 = *(const v8bf*)(qrow + c * 32 + half * 16 + 8);
#pragma unroll
    for (int e = 0; e < 8; ++e) { qa[c][e] = g0[e]; qa[c][8 + e] = g1[e]; }
  }

  v8f o[8];
  float mrow[8], lrow[8];
#pragma unroll
  for (int i = 0; i < 8; ++i) {
#pragma unroll
    for (int e = 0; e < 8; ++e) o[i][e] = 0.0f;
    mrow[i] = -1e30f;
    lrow[i] = 0.0f;
  }

  const bf16* kb0 = kbf + (size_t)b * NT * ND;
  const bf16* vt0 = vtbf + (size_t)b * (size_t)ND * NT;
  int jend = t0 + 15;

  for (int j = 0; j <= jend; j += 32) {
    if (j + 32 <= jend)
      __builtin_prefetch(kb0 + (size_t)(j + 32) * ND, 0, 1);

    // --- scores: S2 = (Q * log2e/sqrt(D)) @ K^T, two 16-wide key tiles ---
    v8f s0, s1;
#pragma unroll
    for (int e = 0; e < 8; ++e) { s0[e] = 0.0f; s1[e] = 0.0f; }
    const bf16* kj = kb0 + (size_t)j * ND;
#pragma unroll
    for (int c = 0; c < 4; ++c) {
      v16bf bk0 = *(const v16bf*)(kj + (size_t)n * ND + c * 32 + half * 16);
      s0 = wmma_bf16(qa[c], bk0, s0);
      v16bf bk1 = *(const v16bf*)(kj + (size_t)(16 + n) * ND + c * 32 + half * 16);
      s1 = wmma_bf16(qa[c], bk1, s1);
    }

    // --- online softmax, base-2 (C layout: row = r+half*8, col = n) ---
    bool diag = (j + 31 > t0);     // wave-uniform: block touches the diagonal
#pragma unroll
    for (int r = 0; r < 8; ++r) {
      float a0, a1;
      if (diag) {
        int trow = t0 + r + half * 8;
        a0 = ((j + n)      <= trow) ? s0[r] : -1e30f;
        a1 = ((j + 16 + n) <= trow) ? s1[r] : -1e30f;
      } else {
        a0 = s0[r];
        a1 = s1[r];
      }
      float mx = fmaxf(a0, a1);
      mx = fmaxf(mx, __shfl_xor(mx, 1));
      mx = fmaxf(mx, __shfl_xor(mx, 2));
      mx = fmaxf(mx, __shfl_xor(mx, 4));
      mx = fmaxf(mx, __shfl_xor(mx, 8));
      float mnew = fmaxf(mrow[r], mx);
      float corr = __builtin_amdgcn_exp2f(mrow[r] - mnew);
      float p0   = __builtin_amdgcn_exp2f(a0 - mnew);
      float p1   = __builtin_amdgcn_exp2f(a1 - mnew);
      float ls = p0 + p1;
      ls += __shfl_xor(ls, 1);
      ls += __shfl_xor(ls, 2);
      ls += __shfl_xor(ls, 4);
      ls += __shfl_xor(ls, 8);
      lrow[r] = lrow[r] * corr + ls;
      mrow[r] = mnew;
#pragma unroll
      for (int dt = 0; dt < 8; ++dt) o[dt][r] *= corr;
      int M = r + half * 8;
      plw[M * 40 + n]      = (bf16)p0;
      plw[M * 40 + 16 + n] = (bf16)p1;
    }

    // wave-internal LDS transpose: wait for DS stores, read back in A layout
    asm volatile("s_wait_dscnt 0x0" ::: "memory");
    v16bf pa;
    {
      v8bf g0 = *(const v8bf*)(&plw[n * 40 + half * 16]);
      v8bf g1 = *(const v8bf*)(&plw[n * 40 + half * 16 + 8]);
#pragma unroll
      for (int e = 0; e < 8; ++e) { pa[e] = g0[e]; pa[8 + e] = g1[e]; }
    }

    // --- O += P @ V (V stored transposed [D, T] -> contiguous B-fragments) ---
    const bf16* vj = vt0 + j;
#pragma unroll
    for (int dt = 0; dt < 8; ++dt) {
      v16bf bv = *(const v16bf*)(vj + (size_t)(dt * 16 + n) * NT + half * 16);
      o[dt] = wmma_bf16(pa, bv, o[dt]);
    }
  }

  // --- normalize + write fp32 output ---
  float* op = out + (size_t)t0g * ND;
#pragma unroll
  for (int r = 0; r < 8; ++r) {
    int M = r + half * 8;
    float inv = 1.0f / lrow[r];
#pragma unroll
    for (int dt = 0; dt < 8; ++dt)
      op[(size_t)M * ND + dt * 16 + n] = o[dt][r] * inv;
  }
}

// ---------------------------------------------------------------------------
extern "C" void kernel_launch(void* const* d_in, const int* in_sizes, int n_in,
                              void* d_out, int out_size, void* d_ws, size_t ws_size,
                              hipStream_t stream) {
  const float* x  = (const float*)d_in[0];
  const float* Wq = (const float*)d_in[1];
  const float* Wk = (const float*)d_in[2];
  const float* Wv = (const float*)d_in[3];
  float* out = (float*)d_out;

  // Workspace layout (bf16 intermediates):
  //   wbf : 3*1024*128        =   786,432 B
  //   qbf : 8*4096*128        = 8,388,608 B
  //   kbf : 8*4096*128        = 8,388,608 B
  //   vtbf: 8*128*4096        = 8,388,608 B   (transposed)
  char* ws = (char*)d_ws;
  bf16* wbf  = (bf16*)(ws);
  bf16* qbf  = (bf16*)(ws + 786432);
  bf16* kbf  = (bf16*)(ws + 786432 + 8388608);
  bf16* vtbf = (bf16*)(ws + 786432 + 2 * 8388608);

  wrefmt_kernel<<<768, 32, 0, stream>>>(Wq, Wk, Wv, wbf);
  qkv_kernel<<<512, 128, 0, stream>>>(x, wbf, qbf, kbf, vtbf);
  attn_kernel<<<512, 128, 0, stream>>>(qbf, kbf, vtbf, out);
}
